// MultiHeadAttention_83923660964585
// MI455X (gfx1250) — matrix operand order repaired and checked by reading, not compiled, not run on hardware
//
#include <hip/hip_runtime.h>
#include <hip/hip_bf16.h>
#include <math.h>

typedef __attribute__((ext_vector_type(16))) _Float16 v16h;
typedef __attribute__((ext_vector_type(2)))  __fp16   v2fp16;
typedef __attribute__((ext_vector_type(8)))  float    v8f;
typedef __attribute__((ext_vector_type(4)))  int      v4i;

#define D_MODEL 1024
#define N_HEAD  8
#define D_HEAD  128
#define L_SEQ   2048
#define BATCH   2
#define ATTN_SCALE 0.03125f   // 1/sqrt(1024)

#define AS1 __attribute__((address_space(1)))
#define AS3 __attribute__((address_space(3)))

#if defined(__has_builtin)
#if __has_builtin(__builtin_amdgcn_global_load_async_to_lds_b128)
#define HAVE_ASYNC_LDS 1
#endif
#endif

union AFrag {
    v16h     v;
    uint32_t u[8];
};

union HexH {           // 16B <-> 8 halves
    uint4    q;
    _Float16 h[8];
};

static __device__ __forceinline__ uint32_t pack_f16x2(float a, float b) {
    union { v2fp16 h; uint32_t u; } cv;
    cv.h = __builtin_amdgcn_cvt_pkrtz(a, b);   // v_cvt_pk_rtz_f16_f32
    return cv.u;
}

// ---------------------------------------------------------------------------
// Projection GEMM: dst[(b*H+h)*L + q][d] = f16( X[b*L+q][:] @ W[:, h*128+d] )
// 256 thr = 8 waves (2 M x 4 N); WG tile 32(M) x 256(N); each wave: 16x64
// strip = 4 C tiles reusing one A fragment (4 wmma per 32-wide K chunk).
// ---------------------------------------------------------------------------
__global__ __launch_bounds__(256) void proj_wmma_kernel(
    const float* __restrict__ X,      // [4096, 1024]
    const float* __restrict__ W,      // [1024, 1024] ([in, out])
    _Float16*    __restrict__ dst)    // [B*H, L, 128] f16
{
    __shared__ _Float16 At[32 * 32];        // [m][k]   2 KB
    __shared__ _Float16 Bt[256 * 32];       // [n][k]  16 KB (W^T tile)

    const int tid  = threadIdx.x;
    const int lane = tid & 31;
    const int wave = tid >> 5;
    const int lhi  = lane >> 4;
    const int lm   = lane & 15;

    const int m0 = (wave >> 2) << 4;        // 0 / 16
    const int n0 = (wave & 3)  << 6;        // 0 / 64 / 128 / 192
    const int tileM = blockIdx.x * 32;
    const int tileN = blockIdx.y * 256;

    v8f acc[4];
#pragma unroll
    for (int t = 0; t < 4; ++t)
        acc[t] = (v8f){0.f, 0.f, 0.f, 0.f, 0.f, 0.f, 0.f, 0.f};

    for (int kc = 0; kc < D_MODEL; kc += 32) {
        // --- Stage A tile: one float4 per thread, packed f16x2 stores -------
        {
            int r = tid >> 3, c = (tid & 7) << 2;
            float4 xv = *(const float4*)(X + (size_t)(tileM + r) * D_MODEL + kc + c);
            uint2 p;
            p.x = pack_f16x2(xv.x, xv.y);
            p.y = pack_f16x2(xv.z, xv.w);
            *(uint2*)(At + r * 32 + c) = p;
        }
        // --- Stage W^T tile: 8 float4 per thread (coalesced along n) -------
#pragma unroll
        for (int it = 0; it < 8; ++it) {
            int cj = tid + it * 256;                // 0..2047 float4 chunks
            int k  = cj >> 6;                       // 0..31
            int n  = (cj & 63) << 2;                // 0..252
            float4 wv = *(const float4*)(W + (size_t)(kc + k) * D_MODEL + tileN + n);
            Bt[(n + 0) * 32 + k] = (_Float16)wv.x;
            Bt[(n + 1) * 32 + k] = (_Float16)wv.y;
            Bt[(n + 2) * 32 + k] = (_Float16)wv.z;
            Bt[(n + 3) * 32 + k] = (_Float16)wv.w;
        }
        __syncthreads();

        AFrag a;
#pragma unroll
        for (int v = 0; v < 8; ++v) {
            int ka = ((v >> 2) << 4) + ((v & 3) << 1) + (lhi << 3);
            a.u[v] = *(const uint32_t*)(At + (m0 + lm) * 32 + ka);
        }
#pragma unroll
        for (int nt = 0; nt < 4; ++nt) {
            AFrag b;
#pragma unroll
            for (int v = 0; v < 8; ++v) {
                int kb = ((v >> 2) << 4) + ((v & 3) << 1) + (lhi << 3);
                b.u[v] = *(const uint32_t*)(Bt + (n0 + nt * 16 + lm) * 32 + kb);
            }
            acc[nt] = __builtin_amdgcn_wmma_f32_16x16x32_f16(
                false, a.v, false, b.v, (short)0, acc[nt], false, false);
        }
        __syncthreads();
    }

    // C layout: VGPR r -> M = r + lhi*8, N = lm
#pragma unroll
    for (int nt = 0; nt < 4; ++nt) {
        const int gc = tileN + n0 + nt * 16 + lm;
        const int h  = gc >> 7;
        const int d  = gc & 127;
#pragma unroll
        for (int r = 0; r < 8; ++r) {
            int gr = tileM + m0 + r + (lhi << 3);
            int bb = gr >> 11;
            int q  = gr & (L_SEQ - 1);
            dst[(((size_t)(bb * N_HEAD + h)) * L_SEQ + q) * D_HEAD + d] = (_Float16)acc[nt][r];
        }
    }
}

// ---------------------------------------------------------------------------
// Flash attention: grid (L/64, B*H), block 128 (4 waves). Wave owns 16 q rows.
// ---------------------------------------------------------------------------
__global__ __launch_bounds__(128) void attn_wmma_kernel(
    const _Float16* __restrict__ Qh,
    const _Float16* __restrict__ Kh,
    const _Float16* __restrict__ Vh,
    const float*    __restrict__ query,
    const unsigned char* __restrict__ mask,
    float*          __restrict__ out)
{
    __shared__ _Float16 Kt[32 * 128];      // [key][d]   8 KB
    __shared__ _Float16 Vt[128 * 32];      // [d][key]   8 KB
    __shared__ _Float16 Pt[4][16 * 32];    // per-wave P staging

    const int tid  = threadIdx.x;
    const int lane = tid & 31;
    const int wave = tid >> 5;
    const int lhi  = lane >> 4;
    const int lm   = lane & 15;

    const int bh = blockIdx.y;
    const int b  = bh >> 3;
    const int h  = bh & 7;
    const int q0 = blockIdx.x * 64 + wave * 16;

    const size_t headBase = (size_t)bh * L_SEQ * D_HEAD;
    const size_t maskBase = (size_t)b * L_SEQ * L_SEQ;

    // Preload Q A-fragments (4 chunks of K=32 over d=0..127)
    AFrag qf[4];
#pragma unroll
    for (int c = 0; c < 4; ++c)
#pragma unroll
        for (int v = 0; v < 8; ++v) {
            int d = c * 32 + ((v >> 2) << 4) + ((v & 3) << 1) + (lhi << 3);
            qf[c].u[v] = *(const uint32_t*)(Qh + headBase + (size_t)(q0 + lm) * D_HEAD + d);
        }

    v8f o[8];
#pragma unroll
    for (int t = 0; t < 8; ++t)
        o[t] = (v8f){0.f, 0.f, 0.f, 0.f, 0.f, 0.f, 0.f, 0.f};
    float mrow[8], lrow[8];
#pragma unroll
    for (int r = 0; r < 8; ++r) { mrow[r] = -INFINITY; lrow[r] = 0.0f; }

    for (int kb = 0; kb < L_SEQ; kb += 32) {
        // ---- Stage K tile [key][d]: CDNA5 async global->LDS if available ----
        const _Float16* ksrc = Kh + headBase + (size_t)kb * D_HEAD;
#if defined(HAVE_ASYNC_LDS)
#pragma unroll
        for (int it = 0; it < 4; ++it) {
            int i = tid + it * 128;                // 512 x b128 transfers
            AS1 void* gv = (AS1 void*)(ksrc + i * 8);
            AS3 void* lv = (AS3 void*)(Kt + i * 8);
            __builtin_amdgcn_global_load_async_to_lds_b128(
                (AS1 v4i*)gv, (AS3 v4i*)lv, 0, 0);
        }
#else
        {
            const uint4* src  = (const uint4*)ksrc;
            uint4*       dstl = (uint4*)Kt;
#pragma unroll
            for (int it = 0; it < 4; ++it) {
                int i = tid + it * 128;
                dstl[i] = src[i];
            }
        }
#endif
        // ---- Stage V tile transposed [d][key]: vector loads, b16 scatter ----
#pragma unroll
        for (int it = 0; it < 4; ++it) {
            int cj  = tid + it * 128;              // 512 uint4 chunks
            int key = cj >> 4;
            int d8  = (cj & 15) << 3;
            HexH vv;
            vv.q = *(const uint4*)(Vh + headBase + (size_t)(kb + key) * D_HEAD + d8);
#pragma unroll
            for (int j = 0; j < 8; ++j)
                Vt[(d8 + j) * 32 + key] = vv.h[j];
        }
        if (kb + 32 < L_SEQ) {
            __builtin_prefetch(Kh + headBase + (size_t)(kb + 32) * D_HEAD, 0, 1);
            __builtin_prefetch(Vh + headBase + (size_t)(kb + 32) * D_HEAD, 0, 1);
        }
#if defined(HAVE_ASYNC_LDS)
#if __has_builtin(__builtin_amdgcn_s_wait_asynccnt)
        __builtin_amdgcn_s_wait_asynccnt(0);
#else
        asm volatile("s_wait_asynccnt 0x0" ::: "memory");
#endif
#endif
        __syncthreads();

        // ---- S = Q K^T : two 16-key column tiles ----
        v8f s[2];
#pragma unroll
        for (int t2 = 0; t2 < 2; ++t2) {
            v8f sacc = (v8f){0.f, 0.f, 0.f, 0.f, 0.f, 0.f, 0.f, 0.f};
#pragma unroll
            for (int c = 0; c < 4; ++c) {
                AFrag bfr;
#pragma unroll
                for (int v = 0; v < 8; ++v) {
                    int d = c * 32 + ((v >> 2) << 4) + ((v & 3) << 1) + (lhi << 3);
                    bfr.u[v] = *(const uint32_t*)(Kt + (t2 * 16 + lm) * 128 + d);
                }
                sacc = __builtin_amdgcn_wmma_f32_16x16x32_f16(
                    false, qf[c].v, false, bfr.v, (short)0, sacc, false, false);
            }
            s[t2] = sacc;
        }

        // ---- scale + mask (True = masked out) ----
#pragma unroll
        for (int t2 = 0; t2 < 2; ++t2)
#pragma unroll
            for (int r = 0; r < 8; ++r) {
                int qrow = q0 + r + (lhi << 3);
                int kcol = kb + t2 * 16 + lm;
                float sv = s[t2][r] * ATTN_SCALE;
                if (mask[maskBase + (size_t)qrow * L_SEQ + kcol]) sv = -INFINITY;
                s[t2][r] = sv;
            }

        // ---- online softmax (row stats across 16-lane halves) ----
        float alpha[8];
#pragma unroll
        for (int r = 0; r < 8; ++r) {
            float bm = fmaxf(s[0][r], s[1][r]);
#pragma unroll
            for (int off = 1; off < 16; off <<= 1)
                bm = fmaxf(bm, __shfl_xor(bm, off, 32));
            float mnew = fmaxf(mrow[r], bm);
            float a_ = (mrow[r] > -INFINITY) ? __expf(mrow[r] - mnew) : 0.0f;
            float p0 = (s[0][r] > -INFINITY) ? __expf(s[0][r] - mnew) : 0.0f;
            float p1 = (s[1][r] > -INFINITY) ? __expf(s[1][r] - mnew) : 0.0f;
            s[0][r] = p0; s[1][r] = p1;
            float rs = p0 + p1;
#pragma unroll
            for (int off = 1; off < 16; off <<= 1)
                rs += __shfl_xor(rs, off, 32);
            lrow[r] = lrow[r] * a_ + rs;
            mrow[r] = mnew;
            alpha[r] = a_;
        }

        // ---- rescale accumulated O ----
#pragma unroll
        for (int t = 0; t < 8; ++t)
#pragma unroll
            for (int r = 0; r < 8; ++r)
                o[t][r] *= alpha[r];

        // ---- C-layout P -> LDS -> A-layout fragment ----
        _Float16* P = Pt[wave];
#pragma unroll
        for (int t2 = 0; t2 < 2; ++t2)
#pragma unroll
            for (int r = 0; r < 8; ++r)
                P[(r + (lhi << 3)) * 32 + t2 * 16 + lm] = (_Float16)s[t2][r];
        AFrag pa;
#pragma unroll
        for (int v = 0; v < 8; ++v) {
            int kk = ((v >> 2) << 4) + ((v & 3) << 1) + (lhi << 3);
            pa.u[v] = *(const uint32_t*)(P + lm * 32 + kk);
        }

        // ---- O += P @ V : 8 d-tiles of 16 ----
#pragma unroll
        for (int t = 0; t < 8; ++t) {
            AFrag bv;
#pragma unroll
            for (int v = 0; v < 8; ++v) {
                int kk = ((v >> 2) << 4) + ((v & 3) << 1) + (lhi << 3);
                bv.u[v] = *(const uint32_t*)(Vt + (t * 16 + lm) * 32 + kk);
            }
            o[t] = __builtin_amdgcn_wmma_f32_16x16x32_f16(
                false, pa.v, false, bv.v, (short)0, o[t], false, false);
        }
        __syncthreads();
    }

    // ---- epilogue: normalize + residual add ----
#pragma unroll
    for (int r = 0; r < 8; ++r) {
        float inv  = (lrow[r] > 0.0f) ? 1.0f / lrow[r] : 0.0f;
        int   qrow = q0 + r + (lhi << 3);
#pragma unroll
        for (int t = 0; t < 8; ++t) {
            int    gcol = h * D_HEAD + t * 16 + lm;
            size_t idx  = ((size_t)b * L_SEQ + qrow) * D_MODEL + gcol;
            out[idx] = o[t][r] * inv + query[idx];
        }
    }
}

// ---------------------------------------------------------------------------
extern "C" void kernel_launch(void* const* d_in, const int* in_sizes, int n_in,
                              void* d_out, int out_size, void* d_ws, size_t ws_size,
                              hipStream_t stream) {
    (void)in_sizes; (void)n_in; (void)out_size; (void)ws_size;
    const float*         query = (const float*)d_in[0];
    const float*         keys  = (const float*)d_in[1];
    const unsigned char* mask  = (const unsigned char*)d_in[2];
    const float*         Wq    = (const float*)d_in[3];
    const float*         Wk    = (const float*)d_in[4];
    const float*         Wv    = (const float*)d_in[5];
    float*               out   = (float*)d_out;

    const size_t headElems = (size_t)BATCH * N_HEAD * L_SEQ * D_HEAD;  // 4M f16 = 8MiB
    _Float16* Qh = (_Float16*)d_ws;
    _Float16* Kh = Qh + headElems;
    _Float16* Vh = Kh + headElems;

    dim3 gp(BATCH * L_SEQ / 32, D_MODEL / 256);  // (128, 4)
    proj_wmma_kernel<<<gp, 256, 0, stream>>>(query, Wq, Qh);
    proj_wmma_kernel<<<gp, 256, 0, stream>>>(keys,  Wk, Kh);
    proj_wmma_kernel<<<gp, 256, 0, stream>>>(keys,  Wv, Vh);

    dim3 ga(L_SEQ / 64, BATCH * N_HEAD);         // (32, 16)
    attn_wmma_kernel<<<ga, 128, 0, stream>>>(Qh, Kh, Vh, query, mask, out);
}
